// MambaBlock_30219389895249
// MI455X (gfx1250) — compile-verified, matrix operand order checked by reading, not run on hardware
//
#include <hip/hip_runtime.h>
#include <hip/hip_bf16.h>

// ---------------------------------------------------------------------------
// Mamba block for MI455X (gfx1250): bf16 WMMA GEMMs (async-LDS double-buffered)
// + f32 selective scan.
//   B=2, L=2048, DIM=1024, D_INNER=2048, D_STATE=16, D_CONV=4, DT_RANK=64
// ---------------------------------------------------------------------------

typedef __attribute__((ext_vector_type(16))) __bf16 v16bf;
typedef __attribute__((ext_vector_type(8)))  float  v8f;

#define BM  128
#define BN  128
#define BKK 32
#define LDT 40   // padded LDS row stride in elements (80 B: 16B-aligned, bank-spread)

// ---------------------------------------------------------------------------
// bf16 WMMA GEMM: C(MxN) = A(MxK) * Bt(NxK)^T, f32 accumulate.
//   A  : M x K row-major bf16
//   Bt : N x K row-major bf16 (i.e. B pre-transposed; N padded to mult of 128)
// mode 0: C = acc
// mode 1: C = res + acc                    (residual epilogue)
// mode 2: C = softplus(acc + bias[col])    (dt epilogue)
// Requirements: M % 128 == 0, N % 128 == 0 (padded), K % 32 == 0.
// Staging is fully async: global_load_async_to_lds_b128 + ASYNCcnt fencing,
// double-buffered so tile k+1 DMA overlaps tile k WMMAs.
// ---------------------------------------------------------------------------
__global__ __launch_bounds__(256)
void gemm_bf16_wmma(const __bf16* __restrict__ A, const __bf16* __restrict__ Bt,
                    float* __restrict__ C, const float* __restrict__ bias,
                    const float* __restrict__ res,
                    int M, int N, int K, int ldc, int mode)
{
    __shared__ __bf16 As[2][BM * LDT];
    __shared__ __bf16 Bs[2][BN * LDT];

    const int tid  = threadIdx.x;
    const int lane = tid & 31;
    const int wave = tid >> 5;
    const int bm0  = blockIdx.y * BM;
    const int bn0  = blockIdx.x * BN;

    const int wm    = (wave >> 1) * 32;      // wave row offset in 128x128 tile
    const int wn    = (wave & 1) * 64;       // wave col offset
    const int lrow  = lane & 15;
    const int khalf = (lane >> 4) << 3;      // 0 or 8 (16-bit A/B VGPR layout)

    // staging map: each thread moves 32 bytes (2 x b128) of A and of Bt
    const int srow  = tid >> 1;              // 0..127
    const int skoff = (tid & 1) << 4;        // 0 or 16 elements

    const __bf16* aptr = A  + (size_t)(bm0 + srow) * K + skoff;
    const __bf16* bptr = Bt + (size_t)(bn0 + srow) * K + skoff;

    const unsigned laA = (unsigned)(size_t)&As[0][srow * LDT + skoff];
    const unsigned laB = (unsigned)(size_t)&Bs[0][srow * LDT + skoff];
    const unsigned bufStride = (unsigned)(BM * LDT * sizeof(__bf16));

    v8f acc[2][4] = {};

    auto issue_tile = [&](int k0, int buf) {
        unsigned la = laA + (unsigned)buf * bufStride;
        unsigned lb = laB + (unsigned)buf * bufStride;
        unsigned long long ga = (unsigned long long)(const void*)(aptr + k0);
        unsigned long long gb = (unsigned long long)(const void*)(bptr + k0);
        asm volatile(
            "global_load_async_to_lds_b128 %0, %2, off\n\t"
            "global_load_async_to_lds_b128 %0, %2, off offset:16\n\t"
            "global_load_async_to_lds_b128 %1, %3, off\n\t"
            "global_load_async_to_lds_b128 %1, %3, off offset:16"
            :: "v"(la), "v"(lb), "v"(ga), "v"(gb) : "memory");
    };

    const int nk = K / BKK;
    issue_tile(0, 0);

    for (int it = 0; it < nk; ++it) {
        const int buf = it & 1;
        asm volatile("s_wait_asynccnt 0x0" ::: "memory");
        __syncthreads();                       // tile `it` visible to all waves

        if (it + 1 < nk)                       // overlap next DMA with WMMAs
            issue_tile((it + 1) * BKK, buf ^ 1);

        v16bf afrag[2], bfrag[4];
#pragma unroll
        for (int m = 0; m < 2; ++m) {
            const __bf16* r = &As[buf][(wm + m * 16 + lrow) * LDT];
            union { uint4 u[2]; v16bf v; } t;
            t.u[0] = *(const uint4*)(r + khalf);        // K = khalf .. +7
            t.u[1] = *(const uint4*)(r + khalf + 16);   // K = khalf+16 .. +23
            afrag[m] = t.v;
        }
#pragma unroll
        for (int n = 0; n < 4; ++n) {
            const __bf16* r = &Bs[buf][(wn + n * 16 + lrow) * LDT];
            union { uint4 u[2]; v16bf v; } t;
            t.u[0] = *(const uint4*)(r + khalf);
            t.u[1] = *(const uint4*)(r + khalf + 16);
            bfrag[n] = t.v;
        }
#pragma unroll
        for (int m = 0; m < 2; ++m)
#pragma unroll
            for (int n = 0; n < 4; ++n)
                acc[m][n] = __builtin_amdgcn_wmma_f32_16x16x32_bf16(
                    false, afrag[m], false, bfrag[n],
                    (short)0, acc[m][n], false, false);
    }

    // epilogue: D layout = VGPR i -> M=i (lanes 0-15) / M=i+8 (lanes 16-31)
    const int mshift = (lane >> 4) << 3;
#pragma unroll
    for (int m = 0; m < 2; ++m) {
#pragma unroll
        for (int n = 0; n < 4; ++n) {
            const int gc = bn0 + wn + n * 16 + (lane & 15);
#pragma unroll
            for (int i = 0; i < 8; ++i) {
                const int gr = bm0 + wm + m * 16 + mshift + i;
                const size_t o = (size_t)gr * ldc + gc;
                const float v = acc[m][n][i];
                if (mode == 1) {
                    C[o] = res[o] + v;
                } else if (mode == 2) {
                    float s = v + bias[gc];
                    C[o] = (s > 20.f) ? s : log1pf(__expf(s));
                } else {
                    C[o] = v;
                }
            }
        }
    }
}

// ---------------------------------------------------------------------------
// Tiled transpose + f32->bf16:  in (K x N, f32) -> out (Npad x K, bf16),
// rows N..Npad-1 zero-filled. Block 32x8, tile 32x32.
// ---------------------------------------------------------------------------
__global__ __launch_bounds__(256)
void transpose_to_bf16(const float* __restrict__ in, __bf16* __restrict__ out,
                       int K, int N, int Npad)
{
    __shared__ float tile[32][33];
    const int kb = blockIdx.y * 32;
    const int nb = blockIdx.x * 32;
    const int tx = threadIdx.x;     // 0..31
    const int ty = threadIdx.y;     // 0..7
#pragma unroll
    for (int r = 0; r < 4; ++r) {
        const int k = kb + ty + r * 8;
        const int n = nb + tx;
        tile[ty + r * 8][tx] = (k < K && n < N) ? in[(size_t)k * N + n] : 0.f;
    }
    __syncthreads();
#pragma unroll
    for (int r = 0; r < 4; ++r) {
        const int n = nb + ty + r * 8;
        const int k = kb + tx;
        if (n < Npad && k < K)
            out[(size_t)n * K + k] = (__bf16)tile[tx][ty + r * 8];
    }
}

// ---------------------------------------------------------------------------
// LayerNorm over last dim (1024) -> bf16. One block per row.
// ---------------------------------------------------------------------------
__global__ __launch_bounds__(256)
void ln_to_bf16(const float* __restrict__ x, const float* __restrict__ gamma,
                const float* __restrict__ beta, __bf16* __restrict__ xn)
{
    __shared__ float s1[256];
    __shared__ float s2[256];
    const int row = blockIdx.x;
    const int tid = threadIdx.x;
    const float* xr = x + (size_t)row * 1024;

    float s = 0.f, q = 0.f;
    for (int j = tid; j < 1024; j += 256) { float v = xr[j]; s += v; q += v * v; }
    s1[tid] = s; s2[tid] = q;
    __syncthreads();
    for (int st = 128; st > 0; st >>= 1) {
        if (tid < st) { s1[tid] += s1[tid + st]; s2[tid] += s2[tid + st]; }
        __syncthreads();
    }
    const float mu  = s1[0] * (1.f / 1024.f);
    const float var = s2[0] * (1.f / 1024.f) - mu * mu;
    const float inv = rsqrtf(var + 1e-5f);
    for (int j = tid; j < 1024; j += 256) {
        float v = (xr[j] - mu) * inv * gamma[j] + beta[j];
        xn[(size_t)row * 1024 + j] = (__bf16)v;
    }
}

// ---------------------------------------------------------------------------
// Extract dt columns (0..63) of proj (stride 128) -> contiguous bf16 (4096x64)
// ---------------------------------------------------------------------------
__global__ void extract_dtp(const float* __restrict__ proj, __bf16* __restrict__ dtp)
{
    int i = blockIdx.x * blockDim.x + threadIdx.x;   // Mr*64
    int r = i >> 6, c = i & 63;
    dtp[i] = (__bf16)proj[(size_t)r * 128 + c];
}

// ---------------------------------------------------------------------------
// Depthwise causal conv (k=4) + bias + SiLU.  xin = xz[:, 0:2048].
// ---------------------------------------------------------------------------
__global__ __launch_bounds__(256)
void conv_silu(const float* __restrict__ xz, const float* __restrict__ cw,
               const float* __restrict__ cb, float* __restrict__ xc,
               __bf16* __restrict__ xcb)
{
    int i = blockIdx.x * blockDim.x + threadIdx.x;   // B*L*D_INNER
    const int e   = i & 2047;
    const int row = i >> 11;          // b*L + t
    const int t   = row & 2047;
    float s = cb[e];
#pragma unroll
    for (int j = 0; j < 4; ++j) {
        const int tt = t - 3 + j;
        if (tt >= 0)
            s += xz[(size_t)(row - 3 + j) * 4096 + e] * cw[e * 4 + j];
    }
    const float r = s * (1.f / (1.f + __expf(-s)));
    xc[i]  = r;
    xcb[i] = (__bf16)r;
}

// ---------------------------------------------------------------------------
// Selective scan: one lane per (b, channel); 16-state h in VGPRs; per-step
// B/C vectors double-buffered in LDS (one barrier per timestep).
// ---------------------------------------------------------------------------
__global__ __launch_bounds__(256)
void scan_kernel(const float* __restrict__ dt, const float* __restrict__ proj,
                 const float* __restrict__ xc, const float* __restrict__ A_log,
                 float* __restrict__ ys)
{
    __shared__ float sBC[2][32];
    const int tid = threadIdx.x;
    const int ch  = blockIdx.x * 256 + tid;   // one b per block (2048 % 256 == 0)
    const int b   = ch >> 11;
    const int e   = ch & 2047;
    const size_t row0 = (size_t)(b << 11);    // b*L

    float Av[16];
#pragma unroll
    for (int n = 0; n < 16; ++n) Av[n] = -__expf(A_log[e * 16 + n]);

    float h[16];
#pragma unroll
    for (int n = 0; n < 16; ++n) h[n] = 0.f;

    if (tid < 32) sBC[0][tid] = proj[row0 * 128 + 64 + tid];
    __syncthreads();

    for (int t = 0; t < 2048; ++t) {
        const int cur = t & 1;
        const size_t row = row0 + t;
        if (tid < 32 && t + 1 < 2048)          // prefetch next step's B/C
            sBC[cur ^ 1][tid] = proj[(row + 1) * 128 + 64 + tid];

        const float dtv = dt[row * 2048 + e];
        const float xv  = xc[row * 2048 + e];
        const float dx  = dtv * xv;
        float y = 0.f;
#pragma unroll
        for (int n = 0; n < 16; ++n) {
            const float dA = __expf(dtv * Av[n]);
            h[n] = dA * h[n] + dx * sBC[cur][n];
            y += h[n] * sBC[cur][16 + n];
        }
        ys[row * 2048 + e] = y;
        __syncthreads();
    }
}

// ---------------------------------------------------------------------------
// y = (ys + xc*Dp) * silu(z)  -> bf16
// ---------------------------------------------------------------------------
__global__ void combine_kernel(const float* __restrict__ ys, const float* __restrict__ xc,
                               const float* __restrict__ xz, const float* __restrict__ Dp,
                               __bf16* __restrict__ yb)
{
    int i = blockIdx.x * blockDim.x + threadIdx.x;
    const int e   = i & 2047;
    const int row = i >> 11;
    const float z = xz[(size_t)row * 4096 + 2048 + e];
    float y = ys[i] + xc[i] * Dp[e];
    y *= z * (1.f / (1.f + __expf(-z)));
    yb[i] = (__bf16)y;
}

// ---------------------------------------------------------------------------
extern "C" void kernel_launch(void* const* d_in, const int* in_sizes, int n_in,
                              void* d_out, int out_size, void* d_ws, size_t ws_size,
                              hipStream_t stream)
{
    (void)in_sizes; (void)n_in; (void)out_size; (void)ws_size;

    const float* x      = (const float*)d_in[0];
    const float* gamma  = (const float*)d_in[1];
    const float* beta   = (const float*)d_in[2];
    const float* W_in   = (const float*)d_in[3];
    const float* conv_w = (const float*)d_in[4];
    const float* conv_b = (const float*)d_in[5];
    const float* W_x    = (const float*)d_in[6];
    const float* W_dt   = (const float*)d_in[7];
    const float* b_dt   = (const float*)d_in[8];
    const float* A_log  = (const float*)d_in[9];
    const float* Dp     = (const float*)d_in[10];
    const float* W_out  = (const float*)d_in[11];
    float* out = (float*)d_out;

    const int Mr = 4096;   // B * L

    char* ws = (char*)d_ws;
    size_t off = 0;
    auto take = [&](size_t bytes) -> char* {
        char* p = ws + off;
        off = (off + bytes + 255) & ~(size_t)255;
        return p;
    };

    __bf16* xn_bf    = (__bf16*)take((size_t)Mr * 1024 * 2);
    __bf16* Wint_bf  = (__bf16*)take((size_t)4096 * 1024 * 2);  // [4096][1024]
    __bf16* Wxt_bf   = (__bf16*)take((size_t)128 * 2048 * 2);   // [128][2048], rows 96.. zero
    __bf16* Wdtt_bf  = (__bf16*)take((size_t)2048 * 64 * 2);    // [2048][64]
    __bf16* Woutt_bf = (__bf16*)take((size_t)1024 * 2048 * 2);  // [1024][2048]
    float*  xz       = (float*) take((size_t)Mr * 4096 * 4);
    float*  xc       = (float*) take((size_t)Mr * 2048 * 4);
    __bf16* xc_bf    = (__bf16*)take((size_t)Mr * 2048 * 2);
    float*  proj     = (float*) take((size_t)Mr * 128 * 4);     // padded N=128
    __bf16* dtp      = (__bf16*)take((size_t)Mr * 64 * 2);
    float*  dtb      = (float*) take((size_t)Mr * 2048 * 4);
    float*  ysb      = (float*) take((size_t)Mr * 2048 * 4);
    __bf16* y_bf     = (__bf16*)take((size_t)Mr * 2048 * 2);

    const dim3 tb(32, 8);

    // weight transpose + bf16 down-convert (Bt layouts for the WMMA GEMMs)
    transpose_to_bf16<<<dim3(4096 / 32, 1024 / 32), tb, 0, stream>>>(W_in,  Wint_bf,  1024, 4096, 4096);
    transpose_to_bf16<<<dim3(128  / 32, 2048 / 32), tb, 0, stream>>>(W_x,   Wxt_bf,   2048,   96,  128);
    transpose_to_bf16<<<dim3(2048 / 32,   64 / 32), tb, 0, stream>>>(W_dt,  Wdtt_bf,    64, 2048, 2048);
    transpose_to_bf16<<<dim3(1024 / 32, 2048 / 32), tb, 0, stream>>>(W_out, Woutt_bf, 2048, 1024, 1024);

    // 1) LayerNorm -> bf16
    ln_to_bf16<<<Mr, 256, 0, stream>>>(x, gamma, beta, xn_bf);

    // 2) xz = xn @ W_in   (4096 x 4096 x 1024)
    gemm_bf16_wmma<<<dim3(4096 / BN, Mr / BM), 256, 0, stream>>>(
        xn_bf, Wint_bf, xz, nullptr, nullptr, Mr, 4096, 1024, 4096, 0);

    // 3) depthwise conv + SiLU
    conv_silu<<<(Mr * 2048) / 256, 256, 0, stream>>>(xz, conv_w, conv_b, xc, xc_bf);

    // 4) proj = xc @ W_x  (4096 x 128pad x 2048)
    gemm_bf16_wmma<<<dim3(128 / BN, Mr / BM), 256, 0, stream>>>(
        xc_bf, Wxt_bf, proj, nullptr, nullptr, Mr, 128, 2048, 128, 0);

    // 5) dt = softplus(proj[:, :64] @ W_dt + b_dt)  (4096 x 2048 x 64)
    extract_dtp<<<(Mr * 64) / 256, 256, 0, stream>>>(proj, dtp);
    gemm_bf16_wmma<<<dim3(2048 / BN, Mr / BM), 256, 0, stream>>>(
        dtp, Wdtt_bf, dtb, b_dt, nullptr, Mr, 2048, 64, 2048, 2);

    // 6) selective scan (sequential over L)
    scan_kernel<<<16, 256, 0, stream>>>(dtb, proj, xc, A_log, ysb);

    // 7) gate: y = (ys + xc*Dp) * silu(z)
    combine_kernel<<<(Mr * 2048) / 256, 256, 0, stream>>>(ysb, xc, xz, Dp, y_bf);

    // 8) out = x + y @ W_out  (4096 x 1024 x 2048), residual epilogue
    gemm_bf16_wmma<<<dim3(1024 / BN, Mr / BM), 256, 0, stream>>>(
        y_bf, Woutt_bf, out, nullptr, x, Mr, 1024, 2048, 1024, 1);
}
